// MLPConv_72198400245773
// MI455X (gfx1250) — matrix operand order repaired and checked by reading, not compiled, not run on hardware
//
#include <hip/hip_runtime.h>
#include <stdint.h>

// ---------------------------------------------------------------------------
// MoNet/GMMConv fused kernel for MI455X (gfx1250, wave32, WMMA 16x16x32 f16)
//
//   N=50000 nodes, E=800000 edges, F_IN=F_OUT=16, DIM=2, HID=128
//   ~79 GFLOP total, dominated by [E,128]@[128,128] and [E,128]@[128,256]
//   with shared weights -> compute bound -> WMMA f16 (f32 accum).
//   Per-edge [16,16] kernels live only in LDS (never hit HBM).
//   Weights are staged into LDS once per workgroup via the Tensor Data
//   Mover (TENSOR_LOAD_TO_LDS, TENSORcnt), overlapped with the gather phase.
// ---------------------------------------------------------------------------

typedef __attribute__((ext_vector_type(16))) _Float16     v16h;
typedef __attribute__((ext_vector_type(8)))  float        v8f;
typedef __attribute__((ext_vector_type(4)))  unsigned int v4u;
typedef __attribute__((ext_vector_type(8)))  int          v8i;
typedef __attribute__((ext_vector_type(4)))  int          v4i;

#define N_NODES 50000
#define N_EDGES 800000
#define F_IN    16
#define F_OUT   16
#define HID     128
#define OUTW    (F_IN * F_OUT)   // 256
#define TILE    64               // edges per workgroup
#define THREADS 256              // 8 wave32s
#define WAVES   8

// Dynamic-LDS layout (bytes); total ~149KB -> 2 WGs per 320KB WGP.
#define SM_W1   0                               // f16 [128][128]  32KB
#define SM_W2   (SM_W1 + HID * HID * 2)         // f16 [256][128]  64KB
#define SM_BIG  (SM_W2 + OUTW * HID * 2)        // f16 [64][256]   32KB (h0 / Wk)
#define SM_H1   (SM_BIG + TILE * OUTW * 2)      // f16 [64][128]   16KB
#define SM_X    (SM_H1 + TILE * HID * 2)        // f32 [64][16]     4KB
#define SM_PS   (SM_X + TILE * F_IN * 4)        // f32 [64][2]     512B
#define SM_SRC  (SM_PS + TILE * 2 * 4)          // i32 [64]        256B
#define SM_DST  (SM_SRC + TILE * 4)             // i32 [64]        256B
#define SM_SIZE (SM_DST + TILE * 4)

// ---------------------------------------------------------------------------
// TDM: async-copy a 2D f16 tile (rows x width, row stride = stride elems)
// from global memory into LDS. D# per CDNA5 ISA.
// Issued once per wave (EXEC ignored); tracked by TENSORcnt.
// This toolchain exposes the 6-arg builtin:
//   (uint32x4 g0, int32x8 g1, int32x4 g2, int32x4 g3, int32x8 g4, i32 cpol)
// ---------------------------------------------------------------------------
__device__ __forceinline__ void tdm_load_f16_2d(const void* gsrc, uint32_t lds_off,
                                                uint32_t width, uint32_t rows,
                                                uint32_t stride) {
  const unsigned long long ga = (unsigned long long)(uintptr_t)gsrc;
  v4u g0;
  g0[0] = 1u;                                     // count=1, user descriptor
  g0[1] = lds_off;                                // lds_addr (bytes)
  g0[2] = (unsigned)(ga & 0xffffffffu);           // global_addr[31:0]
  g0[3] = (unsigned)((ga >> 32) & 0x01ffffffu)    // global_addr[56:32]
        | (2u << 30);                             // type=2 ("image")
  v8i g1;
  g1[0] = (int)(1u << 16);                        // mask=0, data_size=1 (2B)
  g1[1] = (int)((width & 0xffffu) << 16);         // tensor_dim0[15:0]
  g1[2] = (int)(((width >> 16) & 0xffffu)         // tensor_dim0[31:16]
        | ((rows & 0xffffu) << 16));              // tensor_dim1[15:0]
  g1[3] = (int)(((rows >> 16) & 0xffffu)          // tensor_dim1[31:16]
        | ((width & 0xffffu) << 16));             // tile_dim0 = width
  g1[4] = (int)(rows & 0xffffu);                  // tile_dim1 = rows, tile_dim2=0
  g1[5] = (int)stride;                            // tensor_dim0_stride[31:0]
  g1[6] = 0;                                      // stride hi / dim1_stride lo
  g1[7] = 0;
  const v4i z4 = {0, 0, 0, 0};                    // groups 2/3 unused (2D tensor)
  const v8i z8 = {0, 0, 0, 0, 0, 0, 0, 0};
  __builtin_amdgcn_tensor_load_to_lds(g0, g1, z4, z4, z8, 0 /*cpol*/);
}

// ---------------------------------------------------------------------------
// WMMA fragment loaders (ISA 7.12.2 layouts, wave32)
// A (16x32 f16, MxK): lane L holds row M=L&15; v[0..3]:K=k0+{0..7}(+8 upper
// half-wave), v[4..7]:K=k0+{16..23}(+8 upper half-wave), packed lo/hi pairs.
// ---------------------------------------------------------------------------
__device__ __forceinline__ v16h load_a_frag(const _Float16* base,
                                            int ld, int row0, int k0, int lane) {
  const int row = row0 + (lane & 15);
  const int hb  = (lane >> 4) * 8;
  const _Float16* p = base + row * ld + k0;
  v16h a;
#pragma unroll
  for (int j = 0; j < 8; ++j) {
    const int kb = ((j < 4) ? (2 * j) : (16 + 2 * (j - 4))) + hb;
    a[2 * j]     = p[kb];
    a[2 * j + 1] = p[kb + 1];
  }
  return a;
}

// B (32x16 f16, KxN) from TRANSPOSED weights wt[N][K] (ld = K):
// lane L: column N=n0+(L&15); v[0..3]:K=k0+{0..7}(+8 upper half-wave),
// v[4..7]:K=k0+{16..23}(+8 upper half-wave), same order as A.
__device__ __forceinline__ v16h load_b_frag(const _Float16* wt,
                                            int ld, int n0, int k0, int lane) {
  const int col = n0 + (lane & 15);
  const _Float16* p = wt + col * ld + k0 + ((lane >> 4) * 8);
  v16h b;
#pragma unroll
  for (int j = 0; j < 8; ++j) {
    const int kb = (j < 4) ? (2 * j) : (16 + 2 * (j - 4));
    b[2 * j]     = p[kb];
    b[2 * j + 1] = p[kb + 1];
  }
  return b;
}

// ---------------------------------------------------------------------------
// Kernel 0: transpose + convert W1 [128,128], W2 [128,256] to f16 in ws.
//   w1t[n*128+k] = W1[k,n]     w2t[n*128+k] = W2[k,n]
// ---------------------------------------------------------------------------
__global__ void prep_weights_kernel(const float* __restrict__ W1,
                                    const float* __restrict__ W2,
                                    _Float16* __restrict__ w1t,
                                    _Float16* __restrict__ w2t) {
  int idx = blockIdx.x * blockDim.x + threadIdx.x;
  if (idx < HID * HID) {
    int n = idx / HID, k = idx % HID;
    w1t[n * HID + k] = (_Float16)W1[k * HID + n];
  } else if (idx < HID * HID + OUTW * HID) {
    int t = idx - HID * HID;
    int n = t / HID, k = t % HID;
    w2t[n * HID + k] = (_Float16)W2[k * OUTW + n];
  }
}

// ---------------------------------------------------------------------------
// Kernel 1: out[n,o] = bias[o] + sum_i x[n,i]*root_weight[i,o]  (inits d_out)
// ---------------------------------------------------------------------------
__global__ void root_kernel(const float* __restrict__ x,
                            const float* __restrict__ rw,
                            const float* __restrict__ bias,
                            float* __restrict__ out) {
  int idx = blockIdx.x * blockDim.x + threadIdx.x;
  if (idx >= N_NODES * F_OUT) return;
  int n = idx >> 4, o = idx & 15;
  float acc = bias[o];
#pragma unroll
  for (int i = 0; i < F_IN; ++i)
    acc = fmaf(x[n * F_IN + i], rw[i * F_OUT + o], acc);
  out[idx] = acc;
}

// ---------------------------------------------------------------------------
// Kernel 2: fused per-edge MLP (WMMA, weights via TDM) + message + scatter.
// ---------------------------------------------------------------------------
__global__ void __launch_bounds__(THREADS)
edge_kernel(const long long* __restrict__ ei,     // int64 [2,E]
            const float* __restrict__ pseudo,     // [E,2]
            const float* __restrict__ x,          // [N,16]
            const float* __restrict__ W0,         // [2,128]
            const float* __restrict__ b0,         // [128]
            const float* __restrict__ b1,         // [128]
            const float* __restrict__ b2,         // [256]
            const _Float16* __restrict__ w1t,     // [128N,128K] f16 (ws)
            const _Float16* __restrict__ w2t,     // [256N,128K] f16 (ws)
            float* __restrict__ out) {            // [N,16]
  extern __shared__ __align__(16) char smem[];
  _Float16* sh_w1 = (_Float16*)(smem + SM_W1);
  _Float16* sh_w2 = (_Float16*)(smem + SM_W2);
  _Float16* sh_h0 = (_Float16*)(smem + SM_BIG);   // [TILE][HID]
  _Float16* sh_wk = (_Float16*)(smem + SM_BIG);   // [TILE][OUTW], aliases h0
  _Float16* sh_h1 = (_Float16*)(smem + SM_H1);
  float*    sh_x  = (float*)   (smem + SM_X);
  float*    sh_ps = (float*)   (smem + SM_PS);
  int*      sh_src= (int*)     (smem + SM_SRC);
  int*      sh_dst= (int*)     (smem + SM_DST);

  const int tid  = threadIdx.x;
  const int lane = tid & 31;
  const int wave = tid >> 5;
  const int e0   = blockIdx.x * TILE;

  // ---- Kick off TDM weight DMA (wave 0 only; overlaps phases 1+2) --------
  if (wave == 0) {
    tdm_load_f16_2d(w1t, (uint32_t)(uintptr_t)sh_w1, HID, HID,  HID);
    tdm_load_f16_2d(w2t, (uint32_t)(uintptr_t)sh_w2, HID, OUTW, HID);
  }

  // ---- Phase 1: edge indices + pseudo-coords into LDS --------------------
  if (tid < TILE) {
    const long long e = (long long)e0 + tid;
    sh_src[tid] = (e < N_EDGES) ? (int)ei[e] : 0;
  } else if (tid < 2 * TILE) {
    const int t = tid - TILE;
    const long long e = (long long)e0 + t;
    sh_dst[t] = (e < N_EDGES) ? (int)ei[N_EDGES + e] : 0;
  } else {
    const int t = tid - 2 * TILE;           // 0..127
    const long long e = (long long)e0 + (t >> 1);
    sh_ps[t] = (e < N_EDGES) ? pseudo[e * 2 + (t & 1)] : 0.0f;
  }
  // speculative prefetch of next tile's pseudo-coords (global_prefetch_b8)
  if (tid == 0) __builtin_prefetch(pseudo + (size_t)(e0 + TILE) * 2, 0, 1);
  __syncthreads();

  // ---- Phase 2: layer0 (VALU, 2x128) -> f16 h0; gather x[src] ------------
#pragma unroll
  for (int it = 0; it < (TILE * HID) / THREADS; ++it) {
    const int idx = tid + it * THREADS;
    const int e = idx >> 7, j = idx & (HID - 1);
    float v = fmaf(sh_ps[e * 2], W0[j],
              fmaf(sh_ps[e * 2 + 1], W0[HID + j], b0[j]));
    sh_h0[idx] = (_Float16)fmaxf(v, 0.0f);
  }
#pragma unroll
  for (int it = 0; it < (TILE * F_IN) / THREADS; ++it) {
    const int idx = tid + it * THREADS;
    sh_x[idx] = x[sh_src[idx >> 4] * F_IN + (idx & 15)];
  }
  // weight DMA must be complete before anyone reads sh_w1/sh_w2
  if (wave == 0) __builtin_amdgcn_s_wait_tensorcnt(0);
  __syncthreads();

  // ---- Layer 1: h1 = relu(h0 @ W1 + b1)   [64,128]@[128,128], WMMA -------
  for (int t = wave; t < (TILE / 16) * (HID / 16); t += WAVES) {
    const int et = t & 3, ct = t >> 2;
    v8f acc = {};
#pragma unroll
    for (int kc = 0; kc < HID / 32; ++kc) {
      v16h a = load_a_frag(sh_h0, HID, et * 16, kc * 32, lane);
      v16h b = load_b_frag(sh_w1, HID, ct * 16, kc * 32, lane);
      acc = __builtin_amdgcn_wmma_f32_16x16x32_f16(
                false, a, false, b, (short)0, acc, false, false);
    }
    const int col   = ct * 16 + (lane & 15);
    const int rbase = et * 16 + 8 * (lane >> 4);
    const float bv  = b1[col];
#pragma unroll
    for (int r = 0; r < 8; ++r)
      sh_h1[(rbase + r) * HID + col] = (_Float16)fmaxf(acc[r] + bv, 0.0f);
  }
  __syncthreads();   // last read of sh_h0 above; sh_wk aliases it below

  // ---- Layer 2: Wk = h1 @ W2 + b2   [64,128]@[128,256], WMMA -------------
  for (int t = wave; t < (TILE / 16) * (OUTW / 16); t += WAVES) {
    const int et = t & 3, ct = t >> 2;      // ct in 0..15
    v8f acc = {};
#pragma unroll
    for (int kc = 0; kc < HID / 32; ++kc) {
      v16h a = load_a_frag(sh_h1, HID, et * 16, kc * 32, lane);
      v16h b = load_b_frag(sh_w2, HID, ct * 16, kc * 32, lane);
      acc = __builtin_amdgcn_wmma_f32_16x16x32_f16(
                false, a, false, b, (short)0, acc, false, false);
    }
    const int col   = ct * 16 + (lane & 15);
    const int rbase = et * 16 + 8 * (lane >> 4);
    const float bv  = b2[col];
#pragma unroll
    for (int r = 0; r < 8; ++r)
      sh_wk[(rbase + r) * OUTW + col] = (_Float16)(acc[r] + bv);
  }
  __syncthreads();

  // ---- Message + scatter: msg[e,o] = sum_i x[src[e],i]*Wk[e,i,o] ----------
#pragma unroll
  for (int it = 0; it < (TILE * F_OUT) / THREADS; ++it) {
    const int idx = tid + it * THREADS;
    const int e = idx >> 4, o = idx & 15;
    if (e0 + e < N_EDGES) {
      float m = 0.0f;
#pragma unroll
      for (int i = 0; i < F_IN; ++i)
        m = fmaf(sh_x[e * F_IN + i], (float)sh_wk[e * OUTW + i * 16 + o], m);
      atomicAdd(out + sh_dst[e] * F_OUT + o, m);
    }
  }
}

// ---------------------------------------------------------------------------
extern "C" void kernel_launch(void* const* d_in, const int* in_sizes, int n_in,
                              void* d_out, int out_size, void* d_ws, size_t ws_size,
                              hipStream_t stream) {
  const long long* ei  = (const long long*)d_in[0];  // int64 edge_index [2,E]
  const float* pseudo  = (const float*)d_in[1];
  const float* x       = (const float*)d_in[2];
  const float* W0      = (const float*)d_in[3];
  const float* b0      = (const float*)d_in[4];
  const float* W1      = (const float*)d_in[5];
  const float* b1      = (const float*)d_in[6];
  const float* W2      = (const float*)d_in[7];
  const float* b2      = (const float*)d_in[8];
  const float* rw      = (const float*)d_in[9];
  const float* bias    = (const float*)d_in[10];
  float* out           = (float*)d_out;

  // workspace: f16 transposed weights (32KB + 64KB)
  _Float16* w1t = (_Float16*)d_ws;
  _Float16* w2t = w1t + HID * HID;

  {
    const int total = HID * HID + OUTW * HID;          // 49152
    prep_weights_kernel<<<(total + 255) / 256, 256, 0, stream>>>(W1, W2, w1t, w2t);
  }
  {
    const int total = N_NODES * F_OUT;                 // 800000
    root_kernel<<<(total + 255) / 256, 256, 0, stream>>>(x, rw, bias, out);
  }
  {
    const int blocks = (N_EDGES + TILE - 1) / TILE;    // 12500
    edge_kernel<<<blocks, THREADS, SM_SIZE, stream>>>(ei, pseudo, x, W0, b0, b1, b2,
                                                      w1t, w2t, out);
  }
}
